// EEGformer_25615184953817
// MI455X (gfx1250) — matrix operand order repaired and checked by reading, not compiled, not run on hardware
//
#include <hip/hip_runtime.h>
#include <hip/hip_bf16.h>
#include <math.h>

typedef __attribute__((ext_vector_type(16))) __bf16       v16bf;
typedef __attribute__((ext_vector_type(8)))  float        v8f;
typedef __attribute__((ext_vector_type(4)))  unsigned int u32x4;

__device__ __forceinline__ unsigned short f2bf(float f) {
    unsigned int u = __float_as_uint(f);
    u += 0x7FFFu + ((u >> 16) & 1u);           // round-to-nearest-even
    return (unsigned short)(u >> 16);
}
__device__ __forceinline__ float bf2f(unsigned short h) {
    return __uint_as_float(((unsigned int)h) << 16);
}

// ---------------------------------------------------------------------------
// Templated bf16 WMMA GEMM:  C(M,N) = A(M,K) * Bt(N,K)^T  (+ fused epilogue)
//   MODE 0: outF = acc + epi[(row%120)*N + col]      (positional bias, f32)
//   MODE 1: outH = bf16(acc)                          (qkv)
//   MODE 2: outF = acc + epi[row*N + col]             (residual add, f32)
//   MODE 3: outH = bf16(gelu(acc + epi[col]))         (fc1 + exact GELU)
//   MODE 4: outF = acc + epi[col]                     (fc2 + bias, f32)
//
// Tile: 256(M) x 64(N) per 256-thread block, BK=64 staged in LDS.
// 8 waves stacked in M; each wave owns a 32x64 patch = 2(M) x 4(N)
// v_wmma_f32_16x16x32_bf16 accumulators; B fragments broadcast to all waves.
// Per stage: 16 WMMAs for 24 ds_load_b128 + one barrier pair.
// ---------------------------------------------------------------------------
template <int MODE>
__global__ __launch_bounds__(256) void gemm_bf16(
    const unsigned short* __restrict__ A,   // (M,K) bf16 row-major
    const unsigned short* __restrict__ Bt,  // (N,K) bf16 row-major (B transposed)
    float* __restrict__ outF,
    unsigned short* __restrict__ outH,
    const float* __restrict__ epi,
    int M, int N, int K)
{
    (void)M;
    __shared__ __align__(16) unsigned short lA[256 * 72];  // padded stride 72
    __shared__ __align__(16) unsigned short lB[64 * 72];

    const int tid  = threadIdx.x;
    const int lane = tid & 31;
    const int w    = tid >> 5;     // wave id 0..7 -> 32-row slice of M
    const int half = lane >> 4;
    const int l15  = lane & 15;
    const int m0   = blockIdx.y * 256;
    const int n0   = blockIdx.x * 64;

    v8f acc[2][4];
#pragma unroll
    for (int s = 0; s < 2; ++s)
#pragma unroll
        for (int t = 0; t < 4; ++t)
#pragma unroll
            for (int r = 0; r < 8; ++r) acc[s][t][r] = 0.0f;

    const int kTiles = K >> 6;
    for (int kt = 0; kt < kTiles; ++kt) {
        __syncthreads();
        // ---- global -> LDS: A 256x64 (2048 16B vecs), B 64x64 (512 vecs) ----
#pragma unroll
        for (int i = 0; i < 8; ++i) {
            int r = (tid >> 3) + i * 32, c = tid & 7;
            *(u32x4*)(&lA[r * 72 + (c << 3)]) =
                *(const u32x4*)(A + (size_t)(m0 + r) * K + (kt << 6) + (c << 3));
        }
#pragma unroll
        for (int i = 0; i < 2; ++i) {
            int v = tid + i * 256;
            int r = v >> 3, c = v & 7;
            *(u32x4*)(&lB[r * 72 + (c << 3)]) =
                *(const u32x4*)(Bt + (size_t)(n0 + r) * K + (kt << 6) + (c << 3));
        }
        __syncthreads();

        // ---- two K=32 chunks per stage ----
#pragma unroll
        for (int kk = 0; kk < 2; ++kk) {
            union Frag { u32x4 q[2]; v16bf v; };
            Frag af[2], bfr[4];
#pragma unroll
            for (int s = 0; s < 2; ++s) {
                const unsigned short* pa = &lA[(w * 32 + s * 16 + l15) * 72 + kk * 32];
                af[s].q[0] = *(const u32x4*)(pa + half * 8);       // K 8h..8h+7
                af[s].q[1] = *(const u32x4*)(pa + 16 + half * 8);  // K 16+8h..
            }
#pragma unroll
            for (int t = 0; t < 4; ++t) {
                const unsigned short* pb = &lB[(t * 16 + l15) * 72 + kk * 32];
                bfr[t].q[0] = *(const u32x4*)(pb + half * 8);      // K 8h..8h+7
                bfr[t].q[1] = *(const u32x4*)(pb + 16 + half * 8); // K 16+8h..
            }
#pragma unroll
            for (int s = 0; s < 2; ++s)
#pragma unroll
                for (int t = 0; t < 4; ++t)
                    acc[s][t] = __builtin_amdgcn_wmma_f32_16x16x32_bf16(
                        false, af[s].v, false, bfr[t].v, (short)0, acc[s][t], false, false);
        }
    }

    // ---- epilogue: C layout = lane(l15)=N, M = r + 8*half ----
#pragma unroll
    for (int s = 0; s < 2; ++s)
#pragma unroll
        for (int t = 0; t < 4; ++t)
#pragma unroll
            for (int r = 0; r < 8; ++r) {
                int row = m0 + w * 32 + s * 16 + (half << 3) + r;
                int col = n0 + t * 16 + l15;
                float val = acc[s][t][r];
                size_t o = (size_t)row * N + col;
                if (MODE == 0) {
                    val += epi[(size_t)(row % 120) * N + col];
                    outF[o] = val;
                } else if (MODE == 1) {
                    outH[o] = f2bf(val);
                } else if (MODE == 2) {
                    val += epi[o];
                    outF[o] = val;
                } else if (MODE == 3) {
                    val += epi[col];
                    val = 0.5f * val * (1.0f + erff(val * 0.70710678118654752f));
                    outH[o] = f2bf(val);
                } else {  // MODE 4
                    val += epi[col];
                    outF[o] = val;
                }
            }
}

// ---------------------------------------------------------------------------
// Row layernorm over L=960. addx=1 -> out = LN(x)+x. Output bf16.
// ---------------------------------------------------------------------------
__global__ __launch_bounds__(256) void ln_kernel(
    const float* __restrict__ in, const float* __restrict__ g,
    const float* __restrict__ b, unsigned short* __restrict__ out, int addx)
{
    __shared__ float s1[256], s2[256];
    const int row = blockIdx.x, tid = threadIdx.x;
    const float* x = in + (size_t)row * 960;
    float a = 0.f, q = 0.f;
    for (int i = tid; i < 960; i += 256) { float v = x[i]; a += v; q += v * v; }
    s1[tid] = a; s2[tid] = q;
    __syncthreads();
    for (int o = 128; o > 0; o >>= 1) {
        if (tid < o) { s1[tid] += s1[tid + o]; s2[tid] += s2[tid + o]; }
        __syncthreads();
    }
    float m    = s1[0] * (1.0f / 960.0f);
    float var  = s2[0] * (1.0f / 960.0f) - m * m;
    float rstd = rsqrtf(var + 1e-5f);
    for (int i = tid; i < 960; i += 256) {
        float v = x[i];
        float y = (v - m) * rstd * g[i] + b[i];
        if (addx) y += v;
        out[(size_t)row * 960 + i] = f2bf(y);
    }
}

// ---------------------------------------------------------------------------
// Attention: rsa[i,j,h] = (q.k)/sqrt(D); imv = cumsum_j(rsa*v). One wave per
// (i,h): lane reduction via shfl_xor, fp32 running sum, bf16 out.
// qkv: (15360, 2880) bf16, col = x*960 + h*120 + d.  imv: (15360, 960) bf16.
// ---------------------------------------------------------------------------
__global__ __launch_bounds__(32) void attn_kernel(
    const unsigned short* __restrict__ qkv, unsigned short* __restrict__ imv)
{
    const int i = blockIdx.x >> 3;
    const int h = blockIdx.x & 7;
    const int lane = threadIdx.x;
    const float isd = 0.09128709291752768f;  // 1/sqrt(120)
    float acc[4] = {0.f, 0.f, 0.f, 0.f};
    for (int j = 0; j < 120; ++j) {
        size_t base = (size_t)(i * 120 + j) * 2880 + h * 120;
        float p = 0.f;
#pragma unroll
        for (int r = 0; r < 4; ++r) {
            int d = lane + 32 * r;
            if (d < 120) p += bf2f(qkv[base + d]) * bf2f(qkv[base + 960 + d]);
        }
#pragma unroll
        for (int o = 16; o > 0; o >>= 1) p += __shfl_xor(p, o, 32);
        float rsa = p * isd;
        size_t ob = (size_t)(i * 120 + j) * 960 + h * 120;
#pragma unroll
        for (int r = 0; r < 4; ++r) {
            int d = lane + 32 * r;
            if (d < 120) {
                acc[r] += rsa * bf2f(qkv[base + 1920 + d]);
                imv[ob + d] = f2bf(acc[r]);
            }
        }
    }
}

// ---------------------------------------------------------------------------
// Helpers: f32->bf16 convert, transpose-convert, positional-bias table
// ---------------------------------------------------------------------------
__global__ void conv_bf16(const float* __restrict__ in, unsigned short* __restrict__ out, long n)
{
    long i = (long)blockIdx.x * 256 + threadIdx.x;
    long st = (long)gridDim.x * 256;
    for (; i < n; i += st) out[i] = f2bf(in[i]);
}

// in (R,C) f32 -> out (C,R) bf16
__global__ void transp_bf16(const float* __restrict__ in, unsigned short* __restrict__ out, int R, int C)
{
    long n = (long)R * C;
    long i = (long)blockIdx.x * 256 + threadIdx.x;
    long st = (long)gridDim.x * 256;
    for (; i < n; i += st) {
        int c = (int)(i / R), r = (int)(i % R);
        out[i] = f2bf(in[(size_t)r * C + c]);
    }
}

__global__ void bias_kernel(float* __restrict__ bias)
{
    int i = blockIdx.x * 256 + threadIdx.x;
    if (i >= 120 * 960) return;
    int j = i / 960, n = i % 960;
    int t = n >> 1;
    float ang = (float)j * expf(-9.210340371976184f * (4.0f * (float)t) / 960.0f);
    bias[i] = (n & 1) ? cosf(ang) : sinf(ang);
}

// ---------------------------------------------------------------------------
extern "C" void kernel_launch(void* const* d_in, const int* in_sizes, int n_in,
                              void* d_out, int out_size, void* d_ws, size_t ws_size,
                              hipStream_t stream)
{
    (void)in_sizes; (void)n_in; (void)out_size; (void)ws_size;
    const float* x      = (const float*)d_in[0];
    const float* weight = (const float*)d_in[1];
    const float* Wqkv   = (const float*)d_in[2];
    const float* Wo     = (const float*)d_in[3];
    const float* ln_g   = (const float*)d_in[4];
    const float* ln_b   = (const float*)d_in[5];
    const float* lnz_g  = (const float*)d_in[6];
    const float* lnz_b  = (const float*)d_in[7];
    const float* fc1_w  = (const float*)d_in[8];
    const float* fc1_b  = (const float*)d_in[9];
    const float* fc2_w  = (const float*)d_in[10];
    const float* fc2_b  = (const float*)d_in[11];
    float* out = (float*)d_out;

    const int M = 15360, L = 960, N2 = 2880, N4 = 3840;

    char* ws = (char*)d_ws;
    size_t off = 0;
    auto alloc = [&](size_t bytes) -> char* {
        char* p = ws + off;
        off = (off + bytes + 255) & ~(size_t)255;
        return p;
    };
    unsigned short* w_bf    = (unsigned short*)alloc((size_t)L * L * 2);        // weight (n-major = row-major)
    unsigned short* wqkv_bf = (unsigned short*)alloc((size_t)3 * N2 * L * 2);   // (a,2880,960)
    unsigned short* wo_bf   = (unsigned short*)alloc((size_t)3 * L * L * 2);    // (a,960,960)
    unsigned short* fc1t    = (unsigned short*)alloc((size_t)N4 * L * 2);       // (3840,960)
    unsigned short* fc2t    = (unsigned short*)alloc((size_t)L * N4 * 2);       // (960,3840)
    unsigned short* x_bf    = (unsigned short*)alloc((size_t)M * L * 2);
    unsigned short* ln_bf   = (unsigned short*)alloc((size_t)M * L * 2);
    unsigned short* imv_bf  = (unsigned short*)alloc((size_t)M * L * 2);
    unsigned short* s3_bf   = (unsigned short*)alloc((size_t)M * L * 2);
    unsigned short* big     = (unsigned short*)alloc((size_t)M * N4 * 2);       // qkv (M,2880) / h1 (M,3840)
    float*          save2   = (float*)alloc((size_t)M * L * 4);
    float*          bias    = (float*)alloc((size_t)120 * 960 * 4);

    auto gs = [](long n) { long b = (n + 255) / 256; return (unsigned)(b < 4096 ? b : 4096); };

    // Weight / input conversions (bf16, Bt = N-major layouts)
    conv_bf16<<<gs((long)L * L), 256, 0, stream>>>(weight, w_bf, (long)L * L);
    conv_bf16<<<gs((long)3 * N2 * L), 256, 0, stream>>>(Wqkv, wqkv_bf, (long)3 * N2 * L);
    conv_bf16<<<gs((long)3 * L * L), 256, 0, stream>>>(Wo, wo_bf, (long)3 * L * L);
    transp_bf16<<<gs((long)L * N4), 256, 0, stream>>>(fc1_w, fc1t, L, N4);   // -> (3840,960)
    transp_bf16<<<gs((long)N4 * L), 256, 0, stream>>>(fc2_w, fc2t, N4, L);   // -> (960,3840)
    conv_bf16<<<gs((long)M * L), 256, 0, stream>>>(x, x_bf, (long)M * L);
    bias_kernel<<<(120 * 960 + 255) / 256, 256, 0, stream>>>(bias);

    // save = X @ weight^T + posbias            (15360 x 960 x 960)
    gemm_bf16<0><<<dim3(L / 64, M / 256), 256, 0, stream>>>(x_bf, w_bf, out, nullptr, bias, M, L, L);

    for (int a = 0; a < 3; ++a) {
        // ln = LN(save)
        ln_kernel<<<M, 256, 0, stream>>>(out, ln_g, ln_b, ln_bf, 0);
        // qkv = ln @ Wqkv[a]^T                  (15360 x 2880 x 960)
        gemm_bf16<1><<<dim3(N2 / 64, M / 256), 256, 0, stream>>>(
            ln_bf, wqkv_bf + (size_t)a * N2 * L, nullptr, big, nullptr, M, N2, L);
        // rsa + cumsum recurrence
        attn_kernel<<<128 * 8, 32, 0, stream>>>(big, imv_bf);
        // save2 = imv @ Wo[a]^T + save          (15360 x 960 x 960)
        gemm_bf16<2><<<dim3(L / 64, M / 256), 256, 0, stream>>>(
            imv_bf, wo_bf + (size_t)a * L * L, save2, nullptr, out, M, L, L);
        // s3 = LN(save2) + save2
        ln_kernel<<<M, 256, 0, stream>>>(save2, lnz_g, lnz_b, s3_bf, 1);
        // h1 = gelu(s3 @ fc1_w + b1)            (15360 x 3840 x 960)
        gemm_bf16<3><<<dim3(N4 / 64, M / 256), 256, 0, stream>>>(
            s3_bf, fc1t, nullptr, big, fc1_b, M, N4, L);
        // save = h1 @ fc2_w + b2                (15360 x 960 x 3840)
        gemm_bf16<4><<<dim3(L / 64, M / 256), 256, 0, stream>>>(
            big, fc2t, out, nullptr, fc2_b, M, L, N4);
    }
}
